// MultiHeadLinearAttentionWithCache_15375982919766
// MI455X (gfx1250) — matrix operand order repaired and checked by reading, not compiled, not run on hardware
//
#include <hip/hip_runtime.h>
#include <hip/hip_bf16.h>

// ---------------------------------------------------------------------------
// MultiHeadLinearAttentionWithCache for MI455X (gfx1250, wave32, WMMA)
// B=4096, E=1024, H=16, D=64
//
// Roofline: s-state streaming (2.15 GB r+w) dominates at 23.3 TB/s (~92us).
// GEMMs (34 GFLOP) run from L2 via bf16 WMMA (f32 accumulate).
// x / weights / values are converted to bf16 ONCE (streaming pass, ~4us),
// so the GEMM inner loop is pure ds_load_b128 + v_wmma (no VALU conversion)
// and async VMEM->LDS copies (ASYNCcnt) move half the bytes, double-buffered.
// ---------------------------------------------------------------------------

#define Bsz 4096
#define Esz 1024
#define Hsz 16
#define Dsz 64

typedef __attribute__((ext_vector_type(16))) __bf16          v16bf;
typedef __attribute__((ext_vector_type(8)))  float           v8f;
typedef __attribute__((ext_vector_type(4)))  float           f32x4;
typedef __attribute__((ext_vector_type(4)))  unsigned short  u16x4;
typedef __attribute__((ext_vector_type(8)))  unsigned short  v8us;
typedef __attribute__((ext_vector_type(16))) unsigned short  v16us;

// CDNA5 async VMEM->LDS (GLOBAL_LOAD_ASYNC_TO_LDS_B128, ASYNCcnt) if available
#if defined(__has_builtin)
#if __has_builtin(__builtin_amdgcn_global_load_async_to_lds_b128) && \
    __has_builtin(__builtin_amdgcn_s_wait_asynccnt)
#define USE_ASYNC_LDS 1
#endif
#endif

#ifdef USE_ASYNC_LDS
// builtin expects: (v4i addrspace(1)* src, v4i addrspace(3)* dst, imm, imm)
typedef int v4i __attribute__((__vector_size__(16)));
typedef __attribute__((address_space(1))) v4i GV4;
typedef __attribute__((address_space(3))) v4i LV4;
__device__ __forceinline__ void async_copy16(void* lds, const void* g) {
    __builtin_amdgcn_global_load_async_to_lds_b128((GV4*)g, (LV4*)lds, 0, 0);
}
#define ASYNC_WAIT(n) __builtin_amdgcn_s_wait_asynccnt(n)
#else
#define ASYNC_WAIT(n)
#endif

__device__ __forceinline__ unsigned short f2bf(float f) {
    // round-to-nearest-even fp32 -> bf16
    unsigned u = __builtin_bit_cast(unsigned, f);
    unsigned r = u + 0x7FFFu + ((u >> 16) & 1u);
    return (unsigned short)(r >> 16);
}

// ---------------------------------------------------------------------------
// One-shot fp32 -> bf16 streaming conversion (amortizes all GEMM conversions)
// ---------------------------------------------------------------------------
__global__ __launch_bounds__(256)
void cvt_bf16(const float* __restrict__ src, unsigned short* __restrict__ dst)
{
    const size_t i = ((size_t)blockIdx.x * 256 + threadIdx.x) * 4;
    const f32x4 f = __builtin_nontemporal_load((const f32x4*)(src + i));
    u16x4 r;
    #pragma unroll
    for (int j = 0; j < 4; ++j) r[j] = f2bf(f[j]);
    *(u16x4*)(dst + i) = r;
}

// ---------------------------------------------------------------------------
// Tiled GEMM: Out[4096,1024] = act( A[4096,1024]bf16 @ W[1024,1024]bf16 + b )
// Block tile 128x128, K-step 32, 256 threads = 8 waves, wave tile 32x64.
// bf16 tiles double-buffered in LDS via async copies (4 async ops/wave/tile).
// Inner loop: ds_load_b128 fragments + v_wmma_f32_16x16x32_bf16 only.
// gridDim.z selects one of up to 3 (W, bias, Out) sets; z>0 applies elu+1.
// ---------------------------------------------------------------------------
__global__ __launch_bounds__(256)
void proj3_gemm(const unsigned short* __restrict__ Xh,
                const unsigned short* __restrict__ W0, const float* __restrict__ b0, float* __restrict__ O0,
                const unsigned short* __restrict__ W1, const float* __restrict__ b1, float* __restrict__ O1,
                const unsigned short* __restrict__ W2, const float* __restrict__ b2, float* __restrict__ O2)
{
    const unsigned short* W;  const float* bias;  float* Out;  bool act;
    if (blockIdx.z == 0)      { W = W0; bias = b0; Out = O0; act = false; }
    else if (blockIdx.z == 1) { W = W1; bias = b1; Out = O1; act = true;  }
    else                      { W = W2; bias = b2; Out = O2; act = true;  }

    // A: 128 rows x 32 K bf16 (row stride 40 ushort = 80B: 16B-aligned chunks,
    //    bank-spread).  B: staged in fragment order sB[ntile 8][k 32][n 16].
    __shared__ unsigned short sA[2][128][40];    // 2 x 10 KB
    __shared__ unsigned short sB[2][8][32][16];  // 2 x 8 KB

    const int tid  = threadIdx.x;
    const int wave = tid >> 5;
    const int lane = tid & 31;
    const int wm   = wave & 3;           // 4 waves along M (32 rows each)
    const int wn   = wave >> 2;          // 2 waves along N (64 cols each)
    const int gm   = blockIdx.y * 128;
    const int gn   = blockIdx.x * 128;

    v8f acc[2][4] = {};

    // ---- tile copy: 1024 16B-chunks (8 bf16), 4 per thread -> 4 async/wave --
    auto stage = [&](int buf, int k0) {
        #pragma unroll
        for (int i = 0; i < 2; ++i) {
            const int c = tid + i * 256;             // 512 A-chunks
            const int row = c >> 2, cb = (c & 3) << 3;
            const int kc = c >> 4, nc = (c & 15) << 3;   // 512 B-chunks
#ifdef USE_ASYNC_LDS
            async_copy16(&sA[buf][row][cb], Xh + (size_t)(gm + row) * Esz + k0 + cb);
            async_copy16(&sB[buf][nc >> 4][kc][nc & 15],
                         W + (size_t)(k0 + kc) * Esz + gn + nc);
#else
            v8us a = *(const v8us*)(Xh + (size_t)(gm + row) * Esz + k0 + cb);
            v8us b = *(const v8us*)(W + (size_t)(k0 + kc) * Esz + gn + nc);
            *(v8us*)&sA[buf][row][cb] = a;
            *(v8us*)&sB[buf][nc >> 4][kc][nc & 15] = b;
#endif
        }
    };

    stage(0, 0);                                     // preload tile 0

    const int NITER = Esz / 32;                      // 32 K-tiles
    for (int it = 0; it < NITER; ++it) {
        const int cur = it & 1;
        __syncthreads();                             // buf[1-cur] free to overwrite
        if (it + 1 < NITER) {
            stage(1 - cur, (it + 1) * 32);           // DMA next tile (4 async/wave)
            ASYNC_WAIT(4);                           // current tile's ops done
        } else {
            ASYNC_WAIT(0);
        }
        __syncthreads();                             // all waves' tile data in LDS

        // A fragments (ISA 7.12.2 16-bit A 16x32): lane<16 M=lane K{0..7,16..23};
        // lane>=16 M=lane-16 K{8..15,24..31}
        v16bf afrag[2];
        #pragma unroll
        for (int mt = 0; mt < 2; ++mt) {
            const int row = wm * 32 + mt * 16 + (lane & 15);
            const int cb  = (lane >> 4) * 8;
            v8us lo = *(const v8us*)&sA[cur][row][cb];
            v8us hi = *(const v8us*)&sA[cur][row][16 + cb];
            v16us a16 = __builtin_shufflevector(lo, hi,
                0, 1, 2, 3, 4, 5, 6, 7, 8, 9, 10, 11, 12, 13, 14, 15);
            afrag[mt] = __builtin_bit_cast(v16bf, a16);
        }
        #pragma unroll
        for (int nt = 0; nt < 4; ++nt) {
            const int bn = lane & 15, bkb = (lane >> 4) * 8;
            v16us b16;
            #pragma unroll
            for (int j = 0; j < 8; ++j) {
                b16[j]     = sB[cur][wn * 4 + nt][bkb + j][bn];
                b16[8 + j] = sB[cur][wn * 4 + nt][16 + bkb + j][bn];
            }
            v16bf bfrag = __builtin_bit_cast(v16bf, b16);
            #pragma unroll
            for (int mt = 0; mt < 2; ++mt) {
                acc[mt][nt] = __builtin_amdgcn_wmma_f32_16x16x32_bf16(
                    false, afrag[mt], false, bfrag,
                    (short)0, acc[mt][nt], false, false);
            }
        }
    }

    // ---- epilogue: bias (+ optional elu+1) ----
    // C/D layout: lane<16 -> M=r, lane>=16 -> M=8+r ; N = lane&15
    #pragma unroll
    for (int mt = 0; mt < 2; ++mt) {
        #pragma unroll
        for (int nt = 0; nt < 4; ++nt) {
            const int col = gn + wn * 64 + nt * 16 + (lane & 15);
            const float bcol = bias[col];
            #pragma unroll
            for (int r = 0; r < 8; ++r) {
                const int row = gm + wm * 32 + mt * 16 + (lane >> 4) * 8 + r;
                float vv = acc[mt][nt][r] + bcol;
                if (act) vv = (vv > 0.0f) ? (vv + 1.0f) : __expf(vv);  // elu(x)+1
                Out[(size_t)row * Esz + col] = vv;
            }
        }
    }
}

// ---------------------------------------------------------------------------
// Fused state update + normalized read.  16 threads per (b,h) pair, each
// thread owns 4 columns -> b128 non-temporal traffic.  One streaming pass:
//   s_new = s + k (x) v ; z_new = z + k
//   qs[m] = sum_d q[d]*s_new[d][m] ; qz = sum_d q[d]*z_new[d]
//   values[m] = qs[m]/(qz+1e-6)   (emitted directly as bf16 for the out-GEMM)
// ---------------------------------------------------------------------------
__global__ __launch_bounds__(256)
void state_update(const float* __restrict__ s, const float* __restrict__ z,
                  const float* __restrict__ q, const float* __restrict__ k,
                  const float* __restrict__ v,
                  float* __restrict__ s_new, float* __restrict__ z_new,
                  unsigned short* __restrict__ valbf)
{
    __shared__ float sq[16][64], sk[16][64], szk[16][64];   // 12 KB

    const int grp = threadIdx.x >> 4;                 // 0..15 : pair in block
    const int t   = threadIdx.x & 15;                 // 0..15 : 4-col slice
    const size_t p = (size_t)blockIdx.x * 16 + grp;   // (b*H + h)

    const f32x4 q4 = *(const f32x4*)(q + p * Dsz + t * 4);
    const f32x4 k4 = *(const f32x4*)(k + p * Dsz + t * 4);
    const f32x4 z4 = *(const f32x4*)(z + p * Dsz + t * 4);
    const f32x4 zk4 = z4 + k4;                        // z_new slice
    *(f32x4*)&sq[grp][t * 4]  = q4;
    *(f32x4*)&sk[grp][t * 4]  = k4;
    *(f32x4*)&szk[grp][t * 4] = zk4;
    *(f32x4*)(z_new + p * Dsz + t * 4) = zk4;
    __syncthreads();

    const f32x4 vm = *(const f32x4*)(v + p * Dsz + t * 4);
    const float* srow = s     + p * (Dsz * Dsz) + t * 4;
    float*       snro = s_new + p * (Dsz * Dsz) + t * 4;

    f32x4 qs = {0.0f, 0.0f, 0.0f, 0.0f};
    float qz = 0.0f;
    #pragma unroll 8
    for (int d = 0; d < Dsz; ++d) {
        const float kd = sk[grp][d];
        const float qd = sq[grp][d];
        const f32x4 sv = __builtin_nontemporal_load((const f32x4*)(srow + d * Dsz));
        const f32x4 sn = sv + kd * vm;
        __builtin_nontemporal_store(sn, (f32x4*)(snro + d * Dsz));
        qs += qd * sn;
        qz = fmaf(qd, szk[grp][d], qz);
    }

    const float rcp = 1.0f / (qz + 1e-6f);
    u16x4 r;
    #pragma unroll
    for (int j = 0; j < 4; ++j) r[j] = f2bf(qs[j] * rcp);
    *(u16x4*)(valbf + p * Dsz + t * 4) = r;
}

// ---------------------------------------------------------------------------
extern "C" void kernel_launch(void* const* d_in, const int* in_sizes, int n_in,
                              void* d_out, int out_size, void* d_ws, size_t ws_size,
                              hipStream_t stream) {
    const float* x  = (const float*)d_in[0];
    const float* s  = (const float*)d_in[1];
    const float* z  = (const float*)d_in[2];
    const float* Wv = (const float*)d_in[3];
    const float* bv = (const float*)d_in[4];
    const float* Wq = (const float*)d_in[5];
    const float* bq = (const float*)d_in[6];
    const float* Wk = (const float*)d_in[7];
    const float* bk = (const float*)d_in[8];
    const float* Wo = (const float*)d_in[9];
    const float* bo = (const float*)d_in[10];

    const size_t BE = (size_t)Bsz * Esz;                  // 4,194,304
    const size_t EE = (size_t)Esz * Esz;                  // 1,048,576
    float* out   = (float*)d_out;                         // (B,E)
    float* s_new = out + BE;                              // (B,H,D,D)
    float* z_new = s_new + (size_t)Bsz * Hsz * Dsz * Dsz; // (B,H,D)

    // workspace: fp32 q/k/v + bf16 copies of x, values, and the 4 weights
    float* vbuf = (float*)d_ws;                           // 16 MB
    float* qbuf = vbuf + BE;                              // 16 MB (post elu+1)
    float* kbuf = qbuf + BE;                              // 16 MB (post elu+1)
    unsigned short* valbf = (unsigned short*)(kbuf + BE); //  8 MB
    unsigned short* xbf   = valbf + BE;                   //  8 MB
    unsigned short* wvbf  = xbf + BE;                     //  2 MB
    unsigned short* wqbf  = wvbf + EE;                    //  2 MB
    unsigned short* wkbf  = wqbf + EE;                    //  2 MB
    unsigned short* wobf  = wkbf + EE;                    //  2 MB

    // 0) one-shot bf16 conversions (streaming, trivially bandwidth-bound)
    cvt_bf16<<<BE / 1024, 256, 0, stream>>>(x,  xbf);
    cvt_bf16<<<EE / 1024, 256, 0, stream>>>(Wv, wvbf);
    cvt_bf16<<<EE / 1024, 256, 0, stream>>>(Wq, wqbf);
    cvt_bf16<<<EE / 1024, 256, 0, stream>>>(Wk, wkbf);
    cvt_bf16<<<EE / 1024, 256, 0, stream>>>(Wo, wobf);

    // 1) projections: v (no act), q (elu+1), k (elu+1)
    dim3 gproj(Esz / 128, Bsz / 128, 3);
    proj3_gemm<<<gproj, 256, 0, stream>>>(xbf,
                                          wvbf, bv, vbuf,
                                          wqbf, bq, qbuf,
                                          wkbf, bk, kbuf);

    // 2) fused state update + normalized read (streaming, non-temporal)
    state_update<<<(Bsz * Hsz) / 16, 256, 0, stream>>>(s, z, qbuf, kbuf, vbuf,
                                                       s_new, z_new, valbf);

    // 3) output projection: out = values @ Wo + bo (no act) -> d_out
    dim3 gout(Esz / 128, Bsz / 128, 1);
    proj3_gemm<<<gout, 256, 0, stream>>>(valbf,
                                         wobf, bo, out,
                                         wobf, bo, out,
                                         wobf, bo, out);
}
